// LstmFc_30717606101122
// MI455X (gfx1250) — compile-verified, matrix operand order checked by reading, not run on hardware
//
#include <hip/hip_runtime.h>
#include <hip/hip_bf16.h>
#include <hip/hip_fp16.h>

// ---------------------------------------------------------------------------
// LSTM(1->16) -> LSTM(16->16) over T=2048, then FC [1024,32768]x[32768,1000].
//
// Plan (MI455X / gfx1250, wave32):
//  K1: fc_w fp32 -> fp16 into workspace (one pass, 131MB -> 62.5MB).
//  K2: fused 2-layer LSTM. 64 blocks (one per 16-batch tile), 2 waves/block:
//      wave0 = layer0, wave1 = layer1 running one timestep behind (software
//      pipeline through double-buffered LDS h1 tile). Recurrent matmuls use
//      V_WMMA_F32_16X16X4_F32 chained 4x for K=16. Gates activate with
//      v_exp_f32/v_rcp_f32. h2 emitted straight to fp16 feat[B, T*16].
//  K3: GEMM feat(f16) x fc_w(f16)^T with V_WMMA_F32_16X16X32_F16, fp32 acc.
//      Wave computes 32x64 output (2 mtiles x 4 ntiles). Both operands
//      (64MB + 62.5MB fp16) fit the 192MB L2 simultaneously.
//
// WMMA register layouts used (per CDNA5 ISA):
//  f32 A 16x4 : lane L, vgpr j     -> A[m = L%16][k = j + 2*(L/16)]
//  f32 C/D    : lane L, vgpr i     -> D[m = i + 8*(L/16)][n = L%16]
//  f16 A 16x32: lane L, half e     -> A[m = L%16][k = 16*(e/8) + 8*(L/16) + e%8]
//  B layouts assumed to mirror A with M<->N (not verifiable compile-only).
//
// Workspace: fc_w16 (65,536,000 B) then feat16 (67,108,864 B) ~ 126.5 MiB.
// ---------------------------------------------------------------------------

#define T_STEPS 2048
#define HID     16
#define BATCH   1024
#define NCLS    1000
#define KDIM    (T_STEPS * HID)   // 32768
#define LSTRIDE 20                // LDS row stride: bank-spread + 8B aligned

typedef __attribute__((ext_vector_type(2)))  float    v2f;
typedef __attribute__((ext_vector_type(8)))  float    v8f;
typedef __attribute__((ext_vector_type(16))) _Float16 v16h;
typedef __attribute__((ext_vector_type(8)))  _Float16 h8;
typedef __attribute__((ext_vector_type(4)))  _Float16 h4;

__device__ __forceinline__ float fast_rcp(float x) { return __builtin_amdgcn_rcpf(x); }
__device__ __forceinline__ float fast_exp2(float x){ return __builtin_amdgcn_exp2f(x); }
// sigmoid(x) = 1/(1+2^(-x*log2e)),  tanh(x) = 1 - 2/(2^(2x*log2e)+1)
__device__ __forceinline__ float sigm(float x)  { return fast_rcp(1.0f + fast_exp2(-1.4426950408889634f * x)); }
__device__ __forceinline__ float tanh_f(float x){ return 1.0f - 2.0f * fast_rcp(1.0f + fast_exp2(2.8853900817779268f * x)); }

__device__ __forceinline__ v8f wmma_f32x4(v2f a, v2f b, v8f c) {
  return __builtin_amdgcn_wmma_f32_16x16x4_f32(false, a, false, b, (short)0, c, false, false);
}
__device__ __forceinline__ v8f wmma_f16x32(v16h a, v16h b, v8f c) {
  return __builtin_amdgcn_wmma_f32_16x16x32_f16(false, a, false, b, (short)0, c, false, false);
}

// ---------------------------------------------------------------------------
// K1: fp32 -> fp16 weight conversion (vectorized x4)
// ---------------------------------------------------------------------------
__global__ void f32_to_f16_kernel(const float* __restrict__ src,
                                  _Float16* __restrict__ dst, int n4) {
  int i = blockIdx.x * blockDim.x + threadIdx.x;
  if (i < n4) {
    float4 v = ((const float4*)src)[i];
    h4 o = { (_Float16)v.x, (_Float16)v.y, (_Float16)v.z, (_Float16)v.w };
    ((h4*)dst)[i] = o;
  }
}

// ---------------------------------------------------------------------------
// K2: fused 2-layer LSTM, 2-wave pipelined, WMMA f32 16x16x4
// ---------------------------------------------------------------------------
__global__ __launch_bounds__(64) void lstm2_kernel(
    const float* __restrict__ x,     // [B, T]
    const float* __restrict__ Wih0,  // [64, 1]
    const float* __restrict__ Whh0,  // [64, 16]
    const float* __restrict__ bih0, const float* __restrict__ bhh0,
    const float* __restrict__ Wih1,  // [64, 16]
    const float* __restrict__ Whh1,  // [64, 16]
    const float* __restrict__ bih1, const float* __restrict__ bhh1,
    _Float16* __restrict__ feat)     // [B, T*16] fp16
{
  // pipe[s&1]: layer0 output h1 tile [m][k] (also layer0's own recurrent state)
  // st1:       layer1 recurrent state h2 tile [m][k]
  __shared__ __align__(16) float pipe[2][16 * LSTRIDE];
  __shared__ __align__(16) float st1[16 * LSTRIDE];

  const int tid = threadIdx.x;
  const int wid = tid >> 5;          // 0 = layer0 wave, 1 = layer1 wave
  const int l   = tid & 31;
  const int hi  = (l >> 4) & 1;
  const int lm  = l & 15;
  const int b0  = blockIdx.x * 16;   // batch tile base

  for (int i = tid; i < 2 * 16 * LSTRIDE; i += 64) (&pipe[0][0])[i] = 0.0f;
  for (int i = tid; i < 16 * LSTRIDE; i += 64) st1[i] = 0.0f;
  __syncthreads();

  // ---- per-lane weight preload into B-layout register fragments ----------
  // B-frag for chunk g (gate), k-subtile q: value = W[g*16 + lm][q*4 + 2*hi + {0,1}]
  v2f wA[4][4];       // wave0: Whh0     wave1: Wih1
  v2f wB[4][4];       // wave0: unused   wave1: Whh1
  float wx[4];        // wave0: Wih0 column (input is scalar per step)
  float bs[4];        // bias sum per gate, indexed by lane's n=lm
  if (wid == 0) {
#pragma unroll
    for (int g = 0; g < 4; ++g) {
      wx[g] = Wih0[g * 16 + lm];
      bs[g] = bih0[g * 16 + lm] + bhh0[g * 16 + lm];
#pragma unroll
      for (int q = 0; q < 4; ++q) {
        wA[g][q] = *(const v2f*)(Whh0 + (g * 16 + lm) * 16 + q * 4 + 2 * hi);
        wB[g][q] = (v2f){0.0f, 0.0f};
      }
    }
  } else {
#pragma unroll
    for (int g = 0; g < 4; ++g) {
      wx[g] = 0.0f;
      bs[g] = bih1[g * 16 + lm] + bhh1[g * 16 + lm];
#pragma unroll
      for (int q = 0; q < 4; ++q) {
        wA[g][q] = *(const v2f*)(Wih1 + (g * 16 + lm) * 16 + q * 4 + 2 * hi);
        wB[g][q] = *(const v2f*)(Whh1 + (g * 16 + lm) * 16 + q * 4 + 2 * hi);
      }
    }
  }

  v8f c_st = {};                                   // cell state (D layout)
  const float*  xrow = x    + (size_t)(b0 + 8 * hi) * T_STEPS;
  _Float16*     frow = feat + (size_t)(b0 + 8 * hi) * KDIM + lm;

  // Pipeline: iteration s -> wave0 computes layer0 step s (writes pipe[s&1]);
  //           wave1 computes layer1 step s-1 (reads pipe[(s-1)&1]).
  for (int s = 0; s <= T_STEPS; ++s) {
    const int rb = (s + 1) & 1;   // == (s-1)&1 : state/handoff read buffer
    const int wb = s & 1;         // wave0 write buffer

    if (wid == 0) {
      if (s < T_STEPS) {
        const int t = s;
        float xv[8];
#pragma unroll
        for (int i = 0; i < 8; ++i) xv[i] = xrow[(size_t)i * T_STEPS + t];
        v2f hA[4];                                  // h0[t-1] in A layout
#pragma unroll
        for (int q = 0; q < 4; ++q)
          hA[q] = *(const v2f*)(&pipe[rb][lm * LSTRIDE + q * 4 + 2 * hi]);

        v8f acc[4];
#pragma unroll
        for (int g = 0; g < 4; ++g) {
          v8f a;
#pragma unroll
          for (int i = 0; i < 8; ++i) a[i] = xv[i] * wx[g] + bs[g];
#pragma unroll
          for (int q = 0; q < 4; ++q) a = wmma_f32x4(hA[q], wA[g][q], a);
          acc[g] = a;
        }
        v8f hN;
#pragma unroll
        for (int i = 0; i < 8; ++i) {              // gates: i,f,g,o chunks
          float ig = sigm(acc[0][i]);
          float fg = sigm(acc[1][i]);
          float gg = tanh_f(acc[2][i]);
          float og = sigm(acc[3][i]);
          float cn = fg * c_st[i] + ig * gg;
          c_st[i] = cn;
          hN[i] = og * tanh_f(cn);
        }
#pragma unroll
        for (int i = 0; i < 8; ++i)                // D layout -> [m][k] tile
          pipe[wb][(i + 8 * hi) * LSTRIDE + lm] = hN[i];
      }
    } else {
      if (s >= 1) {
        const int t = s - 1;
        v2f h1A[4], h2A[4];
#pragma unroll
        for (int q = 0; q < 4; ++q) {
          h1A[q] = *(const v2f*)(&pipe[rb][lm * LSTRIDE + q * 4 + 2 * hi]);
          h2A[q] = *(const v2f*)(&st1[lm * LSTRIDE + q * 4 + 2 * hi]);
        }
        v8f acc[4];
#pragma unroll
        for (int g = 0; g < 4; ++g) {
          v8f a;
#pragma unroll
          for (int i = 0; i < 8; ++i) a[i] = bs[g];
#pragma unroll
          for (int q = 0; q < 4; ++q) a = wmma_f32x4(h1A[q], wA[g][q], a);
#pragma unroll
          for (int q = 0; q < 4; ++q) a = wmma_f32x4(h2A[q], wB[g][q], a);
          acc[g] = a;
        }
        v8f hN;
#pragma unroll
        for (int i = 0; i < 8; ++i) {
          float ig = sigm(acc[0][i]);
          float fg = sigm(acc[1][i]);
          float gg = tanh_f(acc[2][i]);
          float og = sigm(acc[3][i]);
          float cn = fg * c_st[i] + ig * gg;
          c_st[i] = cn;
          hN[i] = og * tanh_f(cn);
        }
#pragma unroll
        for (int i = 0; i < 8; ++i)
          st1[(i + 8 * hi) * LSTRIDE + lm] = hN[i];
#pragma unroll
        for (int i = 0; i < 8; ++i)                // feat[b, t*16 + n] fp16
          frow[(size_t)i * KDIM + t * HID] = (_Float16)hN[i];
      }
    }
    __syncthreads();
  }
}

// ---------------------------------------------------------------------------
// K3: FC GEMM, fp16 inputs / fp32 accumulate, WMMA 16x16x32.
// Wave: 32 rows x 64 cols. Block: 4 waves (128 rows). Grid: (16 ncol-groups,
// 8 row-groups). A and B both L2-resident.
// ---------------------------------------------------------------------------
__global__ __launch_bounds__(128) void fc_kernel(
    const _Float16* __restrict__ A,   // feat  [1024, 32768]
    const _Float16* __restrict__ Bw,  // fc_w16 [1000, 32768]
    const float* __restrict__ bias,   // [1000]
    float* __restrict__ out)          // [1024, 1000]
{
  const int tid = threadIdx.x;
  const int wid = tid >> 5;
  const int l   = tid & 31;
  const int hi  = (l >> 4) & 1;
  const int lm  = l & 15;

  const int mbase = (blockIdx.y * 4 + wid) * 32;  // 2 mtiles
  const int nbase = blockIdx.x * 64;              // 4 ntiles

  const _Float16* arow[2];
#pragma unroll
  for (int mt = 0; mt < 2; ++mt)
    arow[mt] = A + (size_t)(mbase + mt * 16 + lm) * KDIM + hi * 8;
  const _Float16* brow[4];
#pragma unroll
  for (int nt = 0; nt < 4; ++nt) {
    int r = nbase + nt * 16 + lm;
    if (r > NCLS - 1) r = NCLS - 1;               // clamp tail reads
    brow[nt] = Bw + (size_t)r * KDIM + hi * 8;
  }

  v8f acc[2][4] = {};

  for (int kk = 0; kk < KDIM; kk += 32) {
    v16h af[2], bf[4];
#pragma unroll
    for (int mt = 0; mt < 2; ++mt) {
      h8 lo = *(const h8*)(arow[mt] + kk);        // K = kk + 8*hi + 0..7
      h8 ho = *(const h8*)(arow[mt] + kk + 16);   // K = kk + 16 + 8*hi + 0..7
      af[mt] = __builtin_shufflevector(lo, ho, 0,1,2,3,4,5,6,7,8,9,10,11,12,13,14,15);
    }
#pragma unroll
    for (int nt = 0; nt < 4; ++nt) {
      h8 lo = *(const h8*)(brow[nt] + kk);
      h8 ho = *(const h8*)(brow[nt] + kk + 16);
      bf[nt] = __builtin_shufflevector(lo, ho, 0,1,2,3,4,5,6,7,8,9,10,11,12,13,14,15);
    }
#pragma unroll
    for (int mt = 0; mt < 2; ++mt)
#pragma unroll
      for (int nt = 0; nt < 4; ++nt)
        acc[mt][nt] = wmma_f16x32(af[mt], bf[nt], acc[mt][nt]);
  }

#pragma unroll
  for (int nt = 0; nt < 4; ++nt) {
    int col = nbase + nt * 16 + lm;
    if (col < NCLS) {
      float bv = bias[col];
#pragma unroll
      for (int mt = 0; mt < 2; ++mt)
#pragma unroll
        for (int i = 0; i < 8; ++i) {
          int row = mbase + mt * 16 + i + 8 * hi;
          out[(size_t)row * NCLS + col] = acc[mt][nt][i] + bv;
        }
    }
  }
}

// ---------------------------------------------------------------------------
extern "C" void kernel_launch(void* const* d_in, const int* in_sizes, int n_in,
                              void* d_out, int out_size, void* d_ws, size_t ws_size,
                              hipStream_t stream) {
  const float* x    = (const float*)d_in[0];
  const float* Wih0 = (const float*)d_in[1];
  const float* Whh0 = (const float*)d_in[2];
  const float* bih0 = (const float*)d_in[3];
  const float* bhh0 = (const float*)d_in[4];
  const float* Wih1 = (const float*)d_in[5];
  const float* Whh1 = (const float*)d_in[6];
  const float* bih1 = (const float*)d_in[7];
  const float* bhh1 = (const float*)d_in[8];
  const float* fcw  = (const float*)d_in[9];
  const float* fcb  = (const float*)d_in[10];
  float* out = (float*)d_out;

  // workspace layout: [fc_w fp16 : 1000*32768*2][feat fp16 : 1024*32768*2]
  _Float16* w16  = (_Float16*)d_ws;
  _Float16* feat = (_Float16*)((char*)d_ws + (size_t)NCLS * KDIM * sizeof(_Float16));

  const int n4 = NCLS * KDIM / 4;   // 8,192,000
  f32_to_f16_kernel<<<(n4 + 255) / 256, 256, 0, stream>>>(fcw, w16, n4);
  lstm2_kernel<<<BATCH / 16, 64, 0, stream>>>(x, Wih0, Whh0, bih0, bhh0,
                                              Wih1, Whh1, bih1, bhh1, feat);
  fc_kernel<<<dim3(16, 8), 128, 0, stream>>>(feat, w16, fcb, out);
}
